// DendriticANN_30133490548746
// MI455X (gfx1250) — matrix operand order repaired and checked by reading, not compiled, not run on hardware
//
#include <hip/hip_runtime.h>

#define BDIM   512
#define INDIM  1024
#define HDIM   1024
#define OUTDIM 1000
#define LAYERS 4
#define DDIM   16
#define NEG_SLOPE 0.01f

typedef __attribute__((ext_vector_type(16))) __bf16 v16bf;
typedef __attribute__((ext_vector_type(8)))  __bf16 v8bf;
typedef __attribute__((ext_vector_type(8)))  float  v8f;
typedef __attribute__((ext_vector_type(4)))  float  v4f;
typedef __attribute__((ext_vector_type(2)))  float  v2f;
typedef __attribute__((ext_vector_type(4)))  unsigned int v4u;
typedef int v4i_vs __attribute__((vector_size(16)));   // gcc-style, matches builtin

#ifndef __has_builtin
#define __has_builtin(x) 0
#endif
#if __has_builtin(__builtin_amdgcn_global_load_async_to_lds_b128)
#define HAVE_ASYNC 1
#else
#define HAVE_ASYNC 0
#endif

__device__ __forceinline__ float lrelu(float v) {
    return v >= 0.0f ? v : NEG_SLOPE * v;
}

// 16-byte global -> LDS copy. Async (ASYNCcnt) on gfx1250; sync fallback.
__device__ __forceinline__ void copy16(const void* g, void* l) {
#if HAVE_ASYNC
    typedef __attribute__((address_space(1))) v4i_vs gvec_t;
    typedef __attribute__((address_space(3))) v4i_vs lvec_t;
    __builtin_amdgcn_global_load_async_to_lds_b128((gvec_t*)(g), (lvec_t*)(l), 0, 0);
#else
    *(v4u*)l = *(const v4u*)g;
#endif
}

template <int N>
__device__ __forceinline__ void wait_asynccnt() {
#if HAVE_ASYNC
#if __has_builtin(__builtin_amdgcn_s_wait_asynccnt)
    __builtin_amdgcn_s_wait_asynccnt(N);
#else
    asm volatile("s_wait_asynccnt %0" :: "i"(N));
#endif
#endif
}

// ---- WMMA fragment builders (ISA 7.12.2 layouts) ------------------------

// A (16x32 bf16, MxK) from bf16 LDS. lane%16 = M; lanes<16 K in {0..7,16..23},
// lanes>=16 K in {8..15,24..31}.
__device__ __forceinline__ v16bf frag_a(const __bf16* a_lds, int rbase) {
    const int lane = threadIdx.x & 31;
    const __bf16* p = a_lds + (rbase + (lane & 15)) * 32 + ((lane < 16) ? 0 : 8);
    v8bf lo = *(const v8bf*)(p);
    v8bf hi = *(const v8bf*)(p + 16);
    return __builtin_shufflevector(lo, hi, 0,1,2,3,4,5,6,7,8,9,10,11,12,13,14,15);
}

// A from raw f32 LDS (convert while building fragment).
__device__ __forceinline__ v16bf frag_a(const float* a_lds, int rbase) {
    const int lane = threadIdx.x & 31;
    const float* p = a_lds + (rbase + (lane & 15)) * 32 + ((lane < 16) ? 0 : 8);
    v4f a0 = *(const v4f*)(p);
    v4f a1 = *(const v4f*)(p + 4);
    v4f a2 = *(const v4f*)(p + 16);
    v4f a3 = *(const v4f*)(p + 20);
    v16bf r;
    #pragma unroll
    for (int e = 0; e < 4; ++e) {
        r[e]      = (__bf16)a0[e];
        r[4 + e]  = (__bf16)a1[e];
        r[8 + e]  = (__bf16)a2[e];
        r[12 + e] = (__bf16)a3[e];
    }
    return r;
}

// B (32x16 bf16, KxN) from raw f32 LDS. lane%16 = N; lanes<16 K in {0..7,16..23},
// lanes>=16 K in {8..15,24..31}.
__device__ __forceinline__ v16bf frag_b(const float* b_lds) {
    const int lane = threadIdx.x & 31;
    const float* p = b_lds + (lane & 15) * 32 + ((lane < 16) ? 0 : 8);
    v4f b0 = *(const v4f*)(p);
    v4f b1 = *(const v4f*)(p + 4);
    v4f b2 = *(const v4f*)(p + 16);
    v4f b3 = *(const v4f*)(p + 20);
    v16bf r;
    #pragma unroll
    for (int e = 0; e < 4; ++e) {
        r[e]      = (__bf16)b0[e];
        r[4 + e]  = (__bf16)b1[e];
        r[8 + e]  = (__bf16)b2[e];
        r[12 + e] = (__bf16)b3[e];
    }
    return r;
}

// ---- Generic C[512,N] = act(A[512,K] @ W[N,K]^T + bias) -----------------
template <typename AT, bool ACT, bool OUTBF>
__global__ __launch_bounds__(256)
void gemm_kernel(const AT* __restrict__ A, const float* __restrict__ W,
                 const float* __restrict__ bias, void* __restrict__ Cout,
                 int K, int N) {
    constexpr int EPS = 16 / (int)sizeof(AT);   // elements per 16B segment
    constexpr int SPR = 32 / EPS;               // segments per chunk row
    constexpr int AI  = (BDIM * SPR) / 256;     // segments per thread / per-wave insts

    __shared__ __attribute__((aligned(16))) AT    Abuf[2][BDIM * 32];
    __shared__ __attribute__((aligned(16))) float Bbuf[2][16 * 32];

    const int tid = threadIdx.x, lane = tid & 31, wave = tid >> 5;
    const int n0 = blockIdx.x * 16;
    const int NC = K / 32;

    v8f acc[4];
    #pragma unroll
    for (int i = 0; i < 4; ++i)
        #pragma unroll
        for (int r = 0; r < 8; ++r) acc[i][r] = 0.0f;

    auto stageA = [&](int buf, int k0) {
#if HAVE_ASYNC
        #pragma unroll
        for (int i = 0; i < AI; ++i) {
            int seg = (wave * AI + i) * 32 + lane;
            int row = seg / SPR, s = seg % SPR;
            copy16(A + (size_t)row * K + k0 + s * EPS, &Abuf[buf][row * 32 + s * EPS]);
        }
#else
        v4u tmp[AI];
        #pragma unroll
        for (int i = 0; i < AI; ++i) {           // all loads in flight first
            int seg = (wave * AI + i) * 32 + lane;
            int row = seg / SPR, s = seg % SPR;
            tmp[i] = *(const v4u*)(A + (size_t)row * K + k0 + s * EPS);
        }
        #pragma unroll
        for (int i = 0; i < AI; ++i) {
            int seg = (wave * AI + i) * 32 + lane;
            int row = seg / SPR, s = seg % SPR;
            *(v4u*)&Abuf[buf][row * 32 + s * EPS] = tmp[i];
        }
#endif
    };
    auto stageB = [&](int buf, int k0) {         // 16 x 32 f32, raw copy, rows guarded
        int idx = tid * 2, row = idx >> 5, col = idx & 31;
        int gr = n0 + row;
        v2f f = {0.0f, 0.0f};
        if (gr < N) {
            f = *(const v2f*)(W + (size_t)gr * K + k0 + col);
            if (k0 + 32 < K) __builtin_prefetch(W + (size_t)gr * K + k0 + 32 + col, 0, 1);
        }
        *(v2f*)&Bbuf[buf][row * 32 + col] = f;
    };

    stageA(0, 0);
    stageB(0, 0);
    for (int kc = 0; kc < NC; ++kc) {
        const int nxt = kc + 1;
        if (nxt < NC) {
            stageA(nxt & 1, nxt * 32);
            stageB(nxt & 1, nxt * 32);
            wait_asynccnt<AI>();                 // retire stage kc
        } else {
            wait_asynccnt<0>();
        }
        __syncthreads();
        v16bf bf = frag_b(Bbuf[kc & 1]);
        #pragma unroll
        for (int i = 0; i < 4; ++i) {
            v16bf af = frag_a(&Abuf[kc & 1][0], wave * 64 + i * 16);
            acc[i] = __builtin_amdgcn_wmma_f32_16x16x32_bf16(
                false, af, false, bf, (short)0, acc[i], false, false);
        }
        __syncthreads();
    }

    const int col = n0 + (lane & 15);
    const float bv = (col < N) ? bias[col] : 0.0f;
    const int rb = (lane >= 16) ? 8 : 0;
    #pragma unroll
    for (int i = 0; i < 4; ++i) {
        #pragma unroll
        for (int r = 0; r < 8; ++r) {
            float v = acc[i][r] + bv;
            if (ACT) v = lrelu(v);
            const int row = wave * 64 + i * 16 + r + rb;
            if (col < N) {
                if constexpr (OUTBF) ((__bf16*)Cout)[(size_t)row * N + col] = (__bf16)v;
                else                 ((float*)Cout)[(size_t)row * N + col] = v;
            }
        }
    }
}

// ---- Fused dendrite+soma layer (one neuron per block, Wd streamed once) --
__global__ __launch_bounds__(256)
void dend_soma_kernel(const __bf16* __restrict__ Hin, const float* __restrict__ Wd,
                      const float* __restrict__ sd, __bf16* __restrict__ Hout) {
    constexpr int AI = 8;                        // bf16 A: 8 segs per thread
    __shared__ __attribute__((aligned(16))) __bf16 Abuf[2][BDIM * 32];
    __shared__ __attribute__((aligned(16))) float  Bbuf[2][16 * 32];

    const int tid = threadIdx.x, lane = tid & 31, wave = tid >> 5;
    const int n = blockIdx.x;
    const float* Wn = Wd + (size_t)n * DDIM * HDIM;   // [16 x 1024]
    constexpr int NC = HDIM / 32;

    v8f acc[4];
    #pragma unroll
    for (int i = 0; i < 4; ++i)
        #pragma unroll
        for (int r = 0; r < 8; ++r) acc[i][r] = 0.0f;

    auto stageA = [&](int buf, int k0) {
#if HAVE_ASYNC
        #pragma unroll
        for (int i = 0; i < AI; ++i) {
            int seg = (wave * AI + i) * 32 + lane;
            int row = seg >> 2, s = seg & 3;     // 4 segs per 32-elem row
            copy16(Hin + (size_t)row * HDIM + k0 + s * 8, &Abuf[buf][row * 32 + s * 8]);
        }
#else
        v4u tmp[AI];
        #pragma unroll
        for (int i = 0; i < AI; ++i) {
            int seg = (wave * AI + i) * 32 + lane;
            int row = seg >> 2, s = seg & 3;
            tmp[i] = *(const v4u*)(Hin + (size_t)row * HDIM + k0 + s * 8);
        }
        #pragma unroll
        for (int i = 0; i < AI; ++i) {
            int seg = (wave * AI + i) * 32 + lane;
            int row = seg >> 2, s = seg & 3;
            *(v4u*)&Abuf[buf][row * 32 + s * 8] = tmp[i];
        }
#endif
    };
    auto stageB = [&](int buf, int k0) {
        int idx = tid * 2, row = idx >> 5, col = idx & 31;
        const float* p = Wn + (size_t)row * HDIM + k0 + col;
        *(v2f*)&Bbuf[buf][row * 32 + col] = *(const v2f*)p;
        if (k0 + 32 < HDIM) __builtin_prefetch(p + 32, 0, 1);
    };

    stageA(0, 0);
    stageB(0, 0);
    for (int kc = 0; kc < NC; ++kc) {
        const int nxt = kc + 1;
        if (nxt < NC) {
            stageA(nxt & 1, nxt * 32);
            stageB(nxt & 1, nxt * 32);
            wait_asynccnt<AI>();
        } else {
            wait_asynccnt<0>();
        }
        __syncthreads();
        v16bf bf = frag_b(Bbuf[kc & 1]);
        #pragma unroll
        for (int i = 0; i < 4; ++i) {
            v16bf af = frag_a(&Abuf[kc & 1][0], wave * 64 + i * 16);
            acc[i] = __builtin_amdgcn_wmma_f32_16x16x32_bf16(
                false, af, false, bf, (short)0, acc[i], false, false);
        }
        __syncthreads();
    }

    // Soma: each lane owns d = lane%16, b = rbase + r + (lane>=16 ? 8 : 0).
    const float sdv = sd[n * DDIM + (lane & 15)];
    const int rb = (lane >= 16) ? 8 : 0;
    #pragma unroll
    for (int i = 0; i < 4; ++i) {
        const int rbase = wave * 64 + i * 16;
        #pragma unroll
        for (int r = 0; r < 8; ++r) {
            float t = lrelu(acc[i][r]) * sdv;
            t += __shfl_xor(t, 1);
            t += __shfl_xor(t, 2);
            t += __shfl_xor(t, 4);
            t += __shfl_xor(t, 8);               // 16-lane group reduction over d
            float hv = lrelu(lrelu(t));          // lrelu(soma), then layer lrelu
            if ((lane & 15) == 0)
                Hout[(size_t)(rbase + r + rb) * HDIM + n] = (__bf16)hv;
        }
    }
}

extern "C" void kernel_launch(void* const* d_in, const int* in_sizes, int n_in,
                              void* d_out, int out_size, void* d_ws, size_t ws_size,
                              hipStream_t stream) {
    (void)in_sizes; (void)n_in; (void)out_size; (void)ws_size;
    const float* x     = (const float*)d_in[0];
    const float* W_in  = (const float*)d_in[1];
    const float* b_in  = (const float*)d_in[2];
    const float* Wd    = (const float*)d_in[3];
    const float* sd    = (const float*)d_in[4];
    const float* W_out = (const float*)d_in[5];
    const float* b_out = (const float*)d_in[6];
    float* out = (float*)d_out;

    __bf16* h0 = (__bf16*)d_ws;
    __bf16* h1 = h0 + (size_t)BDIM * HDIM;

    dim3 blk(256);

    // Input layer: h0 = lrelu(x @ W_in^T + b_in), stored bf16
    gemm_kernel<float, true, true><<<HDIM / 16, blk, 0, stream>>>(
        x, W_in, b_in, h0, INDIM, HDIM);

    // Dendritic layers (ping-pong bf16 h)
    const __bf16* cur = h0;
    __bf16* nxt = h1;
    for (int l = 0; l < LAYERS; ++l) {
        dend_soma_kernel<<<HDIM, blk, 0, stream>>>(
            cur, Wd + (size_t)l * HDIM * DDIM * HDIM, sd + (size_t)l * HDIM * DDIM, nxt);
        const __bf16* t = cur; cur = nxt; nxt = (__bf16*)t;
    }

    // Output layer: out = h @ W_out^T + b_out (f32, no activation)
    gemm_kernel<__bf16, false, false><<<(OUTDIM + 15) / 16, blk, 0, stream>>>(
        cur, W_out, b_out, out, HDIM, OUTDIM);
}
